// SharedRoutingObliviousForest_37804302139409
// MI455X (gfx1250) — matrix operand order repaired and checked by reading, not compiled, not run on hardware
//
#include <hip/hip_runtime.h>

// ---------------- problem constants (fixed by setup_inputs) ----------------
#define DIN    1024
#define TTREES 12
#define NDEP   3
#define NCOLS  48            // 36 decision cols + 12 gate cols (3 WMMA N-tiles)
#define LLEAF  8
#define TL     96            // TTREES * LLEAF  (3 WMMA K-steps of 32)
#define HHEADS 3
#define DOUT   512
#define NOUTC  (HHEADS*DOUT) // 1536
#define ROWS   8192          // B*S
#define RB     64            // rows per workgroup
#define CO_STRIDE 104        // padded bf16 coeff row stride (16B-aligned, conflict-free)

typedef __attribute__((ext_vector_type(16))) __bf16          v16bf;
typedef __attribute__((ext_vector_type(16))) unsigned short  v16u;
typedef __attribute__((ext_vector_type(8)))  unsigned short  v8u;
typedef __attribute__((ext_vector_type(8)))  float           v8f;
typedef __attribute__((ext_vector_type(16))) float           v16f;

union BF16x16 { v16u u; v16bf b; };

__device__ __forceinline__ unsigned short f2bf(float f) {
  unsigned int x = __float_as_uint(f);
  x += 0x7FFFu + ((x >> 16) & 1u);          // round-to-nearest-even
  return (unsigned short)(x >> 16);
}

// ---------------- prep kernel: pack phase-1 weights (48 x 1024 bf16) -------
__global__ void pack_wp(const float* __restrict__ dw, const float* __restrict__ gw,
                        unsigned short* __restrict__ wp) {
  int i = blockIdx.x * 256 + threadIdx.x;
  if (i >= NCOLS * DIN) return;
  int row = i / DIN, k = i - row * DIN;
  float v = (row < 36) ? dw[row * DIN + k] : gw[(row - 36) * DIN + k];
  wp[i] = f2bf(v);
}

// ---- prep kernel: pack transposed leaf matrix LpT (1536 x 96 bf16) --------
// LpT[c][k] with c = h*512 + d, k = t*8 + l  <-  leaf_outputs[h][t][l][d]
__global__ void pack_lpt(const float* __restrict__ leaf, unsigned short* __restrict__ lpt) {
  int i = blockIdx.x * 256 + threadIdx.x;
  if (i >= NOUTC * TL) return;
  int c = i / TL, k = i - c * TL;
  int h = c / DOUT, d = c - h * DOUT;
  int t = k / LLEAF, l = k - t * LLEAF;
  lpt[i] = f2bf(leaf[(((size_t)(h * TTREES + t)) * LLEAF + l) * DOUT + d]);
}

// ---------------- fused forest kernel --------------------------------------
__global__ __launch_bounds__(256) void forest_kernel(
    const float* __restrict__ x,            // (8192, 1024)
    const float* __restrict__ dec_b,        // (36,)
    const float* __restrict__ ntl,          // (36,)
    const float* __restrict__ gate_b,       // (12,)
    const unsigned short* __restrict__ wp,  // (48, 1024) bf16
    const unsigned short* __restrict__ lpt, // (1536, 96) bf16
    float* __restrict__ out)                // (3, 8192, 512)
{
  __shared__ float          logits[RB][NCOLS];       // 12 KB
  __shared__ unsigned short coeff[RB * CO_STRIDE];   // 13 KB (bf16)

  const int tid     = threadIdx.x;
  const int w       = tid >> 5;     // wave id 0..7
  const int lane    = tid & 31;
  const int hi      = lane >> 4;    // half-wave select
  const int lm      = lane & 15;
  const int rowBase = blockIdx.x * RB;
  const int mt      = w & 3;        // 16-row M-tile within block

  // ================= Phase 1: logits = X @ [decW ; gateW]^T =================
  {
    const float* xrow = x + (size_t)(rowBase + 16 * mt + lm) * DIN;

    // A fragment builder: lane<16 holds k {kA..kA+15}, kA = k0+hi*16
    auto loadA = [&](int k0) {
      const int kA = k0 + hi * 16;
      const float4* p0 = (const float4*)(xrow + kA);
      const float4* p1 = (const float4*)(xrow + kA + 8);
      float4 a0 = p0[0], a1 = p0[1], a2 = p1[0], a3 = p1[1];
      v16f vf = { a0.x, a0.y, a0.z, a0.w, a1.x, a1.y, a1.z, a1.w,
                  a2.x, a2.y, a2.z, a2.w, a3.x, a3.y, a3.z, a3.w };
      __builtin_prefetch(xrow + kA + 64, 0, 0);      // global_prefetch_b8
      BF16x16 A;
      A.b = __builtin_convertvector(vf, v16bf);      // native packed bf16 cvt
      return A;
    };

    if (w < 4) {                 // waves 0-3: N-tiles 0 and 1, one A build per k
      v8f acc0 = {}, acc1 = {};
      const unsigned short* w0 = wp + (size_t)(0  + lm) * DIN;
      const unsigned short* w1 = wp + (size_t)(16 + lm) * DIN;
      for (int k0 = 0; k0 < DIN; k0 += 32) {
        BF16x16 A = loadA(k0);
        BF16x16 B0, B1;                               // lane<16 -> k0..k0+15
        B0.u = *(const v16u*)(w0 + k0 + hi * 16);
        B1.u = *(const v16u*)(w1 + k0 + hi * 16);
        acc0 = __builtin_amdgcn_wmma_f32_16x16x32_bf16(
                   false, A.b, false, B0.b, (short)0, acc0, false, false);
        acc1 = __builtin_amdgcn_wmma_f32_16x16x32_bf16(
                   false, A.b, false, B1.b, (short)0, acc1, false, false);
      }
      #pragma unroll
      for (int j = 0; j < 8; ++j) {                  // D: VGPR j -> M=j (lo)/j+8 (hi)
        logits[16 * mt + j + hi * 8][ 0 + lm] = acc0[j];
        logits[16 * mt + j + hi * 8][16 + lm] = acc1[j];
      }
    } else {                     // waves 4-7: N-tile 2
      v8f acc = {};
      const unsigned short* w2 = wp + (size_t)(32 + lm) * DIN;
      for (int k0 = 0; k0 < DIN; k0 += 32) {
        BF16x16 A = loadA(k0);
        BF16x16 Bf;
        Bf.u = *(const v16u*)(w2 + k0 + hi * 16);
        acc = __builtin_amdgcn_wmma_f32_16x16x32_bf16(
                  false, A.b, false, Bf.b, (short)0, acc, false, false);
      }
      #pragma unroll
      for (int j = 0; j < 8; ++j)
        logits[16 * mt + j + hi * 8][32 + lm] = acc[j];
    }
  }
  __syncthreads();

  // ====== per-row: sigmoid routing, leaf probs, gate softmax -> coeff ======
  {
    const int rl = tid >> 2;   // 0..63 local row
    const int q  = tid & 3;    // quarter of trees (3 trees each)
    float g[TTREES];
    float gmax = -3.0e38f;
    #pragma unroll
    for (int t = 0; t < TTREES; ++t) {
      g[t] = logits[rl][36 + t] + gate_b[t];
      gmax = fmaxf(gmax, g[t]);
    }
    float gsum = 0.f;
    #pragma unroll
    for (int t = 0; t < TTREES; ++t) { g[t] = __expf(g[t] - gmax); gsum += g[t]; }
    const float ginv = 1.f / gsum;
    #pragma unroll
    for (int ti = 0; ti < 3; ++ti) {
      const int t = q * 3 + ti;
      const float wgt = g[t] * ginv;
      float d[NDEP];
      #pragma unroll
      for (int n = 0; n < NDEP; ++n) {
        const int idx = t * NDEP + n;
        const float temp = log1pf(__expf(ntl[idx] + 0.5413f));  // TEMPERATURE==1
        const float z = (logits[rl][idx] + dec_b[idx]) / temp;
        d[n] = 1.f / (1.f + __expf(-z));
      }
      #pragma unroll
      for (int l = 0; l < LLEAF; ++l) {
        const float p = ((l & 4) ? 1.f - d[0] : d[0]) *
                        ((l & 2) ? 1.f - d[1] : d[1]) *
                        ((l & 1) ? 1.f - d[2] : d[2]);
        coeff[rl * CO_STRIDE + t * LLEAF + l] = f2bf(p * wgt);
      }
    }
  }
  __syncthreads();

  // ============ Phase 2: out = coeff(64x96) @ LpT^T(96x1536) ===============
  {
    const int mt2 = w & 3;       // M-tile
    const int nh  = w >> 2;      // N-half: 48 N-tiles each
    const unsigned short* crow = coeff + (size_t)(16 * mt2 + lm) * CO_STRIDE;
    BF16x16 A[3];
    #pragma unroll
    for (int ks = 0; ks < 3; ++ks) {            // cache full-K A fragments
      const int kb = 32 * ks + hi * 16;
      v8u lo = *(const v8u*)(crow + kb);
      v8u hh = *(const v8u*)(crow + kb + 8);
      #pragma unroll
      for (int e = 0; e < 8; ++e) { A[ks].u[e] = lo[e]; A[ks].u[e + 8] = hh[e]; }
    }
    for (int j = 0; j < 48; ++j) {
      const int colBase = 16 * (nh * 48 + j);
      const unsigned short* brow = lpt + (size_t)(colBase + lm) * TL;
      v8f acc = {};
      #pragma unroll
      for (int ks = 0; ks < 3; ++ks) {
        BF16x16 Bf;
        Bf.u = *(const v16u*)(brow + 32 * ks + hi * 16);
        acc = __builtin_amdgcn_wmma_f32_16x16x32_bf16(
                  false, A[ks].b, false, Bf.b, (short)0, acc, false, false);
      }
      const int h  = colBase >> 9;        // / 512
      const int d0 = colBase & 511;
      float* op = out + (size_t)h * ROWS * DOUT;
      #pragma unroll
      for (int j2 = 0; j2 < 8; ++j2) {
        const int row = rowBase + 16 * mt2 + j2 + hi * 8;
        op[(size_t)row * DOUT + d0 + lm] = acc[j2];
      }
    }
  }
}

// ---------------------------------------------------------------------------
extern "C" void kernel_launch(void* const* d_in, const int* in_sizes, int n_in,
                              void* d_out, int out_size, void* d_ws, size_t ws_size,
                              hipStream_t stream) {
  (void)in_sizes; (void)n_in; (void)out_size; (void)ws_size;
  const float* x    = (const float*)d_in[0];
  const float* dw   = (const float*)d_in[1];  // (12,3,1024)
  const float* db   = (const float*)d_in[2];  // (12,3)
  const float* ntl  = (const float*)d_in[3];  // (12,3)
  const float* gw   = (const float*)d_in[4];  // (12,1024)
  const float* gb   = (const float*)d_in[5];  // (12,)
  const float* leaf = (const float*)d_in[6];  // (3,12,8,512)
  float* out = (float*)d_out;

  unsigned short* wp  = (unsigned short*)d_ws;          // 48*1024 bf16 = 96 KB
  unsigned short* lpt = wp + NCOLS * DIN;               // 1536*96 bf16 = 288 KB

  pack_wp <<< (NCOLS * DIN + 255) / 256, 256, 0, stream >>>(dw, gw, wp);
  pack_lpt<<< (NOUTC * TL  + 255) / 256, 256, 0, stream >>>(leaf, lpt);
  forest_kernel<<< ROWS / RB, 256, 0, stream >>>(x, db, ntl, gb, wp, lpt, out);
}
